// DynamicConv_80187039416867
// MI455X (gfx1250) — matrix operand order repaired and checked by reading, not compiled, not run on hardware
//
#include <hip/hip_runtime.h>
#include <cstdint>
#include <cstddef>

// ---------- vector types matching CDNA5 WMMA operand shapes ----------
typedef __attribute__((ext_vector_type(16))) __bf16          v16bf;
typedef __attribute__((ext_vector_type(8)))  float           v8f;
typedef __attribute__((ext_vector_type(8)))  unsigned short  u16x8;
typedef __attribute__((ext_vector_type(16))) unsigned short  u16x16;

// Problem constants (fixed by the reference)
#define BN   32
#define CN   256
#define ON   256
#define HWN  3136     // 56*56
#define WIMG 56
#define KKN  2304     // C * 3 * 3
#define EN   8

#define LSTR 40       // LDS row stride in ushorts (80B): 16B-aligned, bank-conflict-free

// fp32 -> bf16 (round to nearest even, NaN not special-cased)
static __device__ __forceinline__ unsigned short f2bf(float f) {
    unsigned int u = __float_as_uint(f);
    u += 0x7FFFu + ((u >> 16) & 1u);
    return (unsigned short)(u >> 16);
}

// ---------------- Kernel 1: global average pool over H,W ----------------
__global__ __launch_bounds__(256) void pool_kernel(const float* __restrict__ x,
                                                   float* __restrict__ pooled) {
    const int bc = blockIdx.x;                       // 0 .. B*C-1
    const float* p = x + (size_t)bc * HWN;
    float s = 0.0f;
    for (int i = threadIdx.x; i < HWN; i += 256) s += p[i];
    __shared__ float red[256];
    red[threadIdx.x] = s;
    __syncthreads();
    for (int off = 128; off > 0; off >>= 1) {
        if ((int)threadIdx.x < off) red[threadIdx.x] += red[threadIdx.x + off];
        __syncthreads();
    }
    if (threadIdx.x == 0) pooled[bc] = red[0] * (1.0f / (float)HWN);
}

// ---------------- Kernel 2: routing MLP + sigmoid + group-mean ----------------
__global__ __launch_bounds__(64) void route_kernel(const float* __restrict__ pooled,
                                                   const float* __restrict__ pw,
                                                   const float* __restrict__ pb,
                                                   float* __restrict__ r) {
    const int b = blockIdx.x;
    const int d = threadIdx.x;                        // 0..63
    const float* pv = pooled + b * CN;
    const float* wv = pw + d * CN;
    float acc = pb[d];
    for (int c = 0; c < CN; ++c) acc += pv[c] * wv[c];
    const float s = 1.0f / (1.0f + expf(-acc));
    __shared__ float sig[64];
    sig[d] = s;
    __syncthreads();
    if (d < EN) {
        float m = 0.0f;
        #pragma unroll
        for (int i = 0; i < 8; ++i) m += sig[d * 8 + i];
        r[b * EN + d] = m * 0.125f;
    }
}

// ---------------- Kernel 3: expert mix -> bf16 weights, K-permuted ----------------
// Output layout: wmix[b][o][k'] with k' = tap*256 + c   (tap = ky*3+kx)
__global__ __launch_bounds__(256) void mix_kernel(const float* __restrict__ expert_w,
                                                  const float* __restrict__ r,
                                                  unsigned short* __restrict__ wmix) {
    const int o = blockIdx.x;                         // 0..255
    __shared__ float rs[BN * EN];                     // 256 routing weights
    for (int i = threadIdx.x; i < BN * EN; i += 256) rs[i] = r[i];
    __syncthreads();

    for (int kp = threadIdx.x; kp < KKN; kp += 256) {
        const int c = kp & 255;                       // channel
        const int t = kp >> 8;                        // tap 0..8
        float w[EN];
        #pragma unroll
        for (int e = 0; e < EN; ++e)
            w[e] = expert_w[(size_t)(e * ON + o) * KKN + (c * 9 + t)];
        #pragma unroll
        for (int b = 0; b < BN; ++b) {
            float s = 0.0f;
            #pragma unroll
            for (int e = 0; e < EN; ++e) s += rs[b * EN + e] * w[e];
            wmix[((size_t)b * ON + o) * KKN + kp] = f2bf(s);
        }
    }
}

// ---------------- Kernel 4: implicit-GEMM conv via v_wmma_f32_16x16x32_bf16 ----------------
// Workgroup: (b, 128 o-channels, 64 spatial positions). 8 waves = 4(M) x 2(N),
// each wave computes 32x32 (4 accumulators; 2 A frags x 2 B frags).
// K permuted tap-major; double-buffered LDS software pipeline:
//   iteration s: issue loads for s+1  ->  4 WMMAs on stage s  ->  store gathered s+1.
__global__ __launch_bounds__(256) void conv_kernel(const float* __restrict__ x,
                                                   const unsigned short* __restrict__ wmix,
                                                   float* __restrict__ out) {
    __shared__ unsigned short ldsA[2][128 * LSTR];    // weights tile [o_row][c]
    __shared__ unsigned short ldsB[2][64 * LSTR];     // im2col tile  [n][c]

    const int b    = blockIdx.z;
    const int o0   = blockIdx.y * 128;
    const int pos0 = blockIdx.x * 64;
    const int tid  = threadIdx.x;
    const int lane = tid & 31;
    const int mw   = (tid >> 5) >> 1;                 // 0..3 : M sub-tile (32 rows)
    const int nw   = (tid >> 5) & 1;                  // 0..1 : N sub-tile (32 wide)

    // ---- A loader: 128 rows x 32 c; two async b128 per thread per stage ----
    const int ar = tid >> 1;                          // 0..127
    const int ak = (tid & 1) * 16;                    // 0 or 16
    const unsigned short* aRow = wmix + (size_t)(b * ON + o0 + ar) * KKN + ak;
    const unsigned aLds[2][2] = {
        { (unsigned)(uintptr_t)&ldsA[0][ar * LSTR + ak],
          (unsigned)(uintptr_t)&ldsA[0][ar * LSTR + ak + 8] },
        { (unsigned)(uintptr_t)&ldsA[1][ar * LSTR + ak],
          (unsigned)(uintptr_t)&ldsA[1][ar * LSTR + ak + 8] } };

    // ---- B loader: 64 n x 32 c; thread -> channel pair h x n-base bn0, 4 n's ----
    const int h   = tid & 15;                         // c = c0 + 2h, 2h+1
    const int bn0 = tid >> 4;                         // n = bn0 + 16*j
    const float* xbase = x + (size_t)b * CN * HWN;

    int py[4], px[4];
    #pragma unroll
    for (int j = 0; j < 4; ++j) {
        const int pos = pos0 + bn0 + 16 * j;
        py[j] = pos / WIMG;
        px[j] = pos - py[j] * WIMG;
    }

    int      off[4];
    unsigned okm[4];                                  // all-ones / zero VGPR masks
    auto setTap = [&](int t) {
        const int dy = t / 3 - 1;
        const int dx = t - (t / 3) * 3 - 1;
        #pragma unroll
        for (int j = 0; j < 4; ++j) {
            const int iy = py[j] + dy;
            const int ix = px[j] + dx;
            const bool v = ((unsigned)iy < (unsigned)WIMG) & ((unsigned)ix < (unsigned)WIMG);
            okm[j] = v ? ~0u : 0u;
            off[j] = (iy * WIMG + ix) & (int)okm[j];  // clamped safe address
        }
    };

    // ---------- prologue: stage 0 (tap 0, c0 = 0) into buffer 0 ----------
    setTap(0);
    asm volatile("global_load_async_to_lds_b128 %0, %1, off"
                 :: "v"(aLds[0][0]), "v"(aRow) : "memory");
    asm volatile("global_load_async_to_lds_b128 %0, %1, off"
                 :: "v"(aLds[0][1]), "v"(aRow + 8) : "memory");
    {
        const float* xc0 = xbase + (size_t)(2 * h) * HWN;
        const float* xc1 = xc0 + HWN;
        #pragma unroll
        for (int j = 0; j < 4; ++j) {
            const unsigned pk = ((unsigned)f2bf(xc0[off[j]]) |
                                 ((unsigned)f2bf(xc1[off[j]]) << 16)) & okm[j];
            *(unsigned*)&ldsB[0][(bn0 + 16 * j) * LSTR + 2 * h] = pk;
        }
    }
    asm volatile("s_wait_asynccnt 0x0" ::: "memory");
    __syncthreads();

    v8f acc00 = {}, acc01 = {}, acc10 = {}, acc11 = {};

    for (int s = 0; s < 72; ++s) {                    // 72 stages of K=32
        const int cur = s & 1;
        const unsigned short* At = ldsA[cur];
        const unsigned short* Bt = ldsB[cur];

        // ---- issue loads for stage s+1 (other buffer) ----
        const int  sl = s + 1;
        const bool haveNext = (sl < 72);
        float vals[8];
        if (haveNext) {
            const int tl = sl >> 3;
            const int cl = (sl & 7) * 32;
            if ((sl & 7) == 0) setTap(tl);
            const unsigned short* ga = aRow + tl * 256 + cl;
            asm volatile("global_load_async_to_lds_b128 %0, %1, off"
                         :: "v"(aLds[sl & 1][0]), "v"(ga) : "memory");
            asm volatile("global_load_async_to_lds_b128 %0, %1, off"
                         :: "v"(aLds[sl & 1][1]), "v"(ga + 8) : "memory");
            const float* xc0 = xbase + (size_t)(cl + 2 * h) * HWN;
            const float* xc1 = xc0 + HWN;
            #pragma unroll
            for (int j = 0; j < 4; ++j) {
                vals[2 * j]     = xc0[off[j]];
                vals[2 * j + 1] = xc1[off[j]];
            }
        }

        // ---- compute stage s: fragments per ISA VGPR layouts + 4 WMMAs ----
        const int am  = mw * 32 + (lane & 15);
        const int alo = (lane < 16) ? 0 : 8;
        u16x16 avu0, avu1;
        {
            const u16x8 a0v = *(const u16x8*)&At[am * LSTR + alo];
            const u16x8 a1v = *(const u16x8*)&At[am * LSTR + alo + 16];
            const u16x8 a2v = *(const u16x8*)&At[(am + 16) * LSTR + alo];
            const u16x8 a3v = *(const u16x8*)&At[(am + 16) * LSTR + alo + 16];
            #pragma unroll
            for (int i = 0; i < 8; ++i) {
                avu0[i] = a0v[i]; avu0[i + 8] = a1v[i];
                avu1[i] = a2v[i]; avu1[i + 8] = a3v[i];
            }
        }
        const int bnn = nw * 32 + (lane & 15);
        const int blo = (lane < 16) ? 0 : 8;
        u16x16 bvu0, bvu1;
        {
            const u16x8 b0 = *(const u16x8*)&Bt[bnn * LSTR + blo];
            const u16x8 b1 = *(const u16x8*)&Bt[bnn * LSTR + blo + 16];
            const u16x8 b2 = *(const u16x8*)&Bt[(bnn + 16) * LSTR + blo];
            const u16x8 b3 = *(const u16x8*)&Bt[(bnn + 16) * LSTR + blo + 16];
            #pragma unroll
            for (int i = 0; i < 8; ++i) {
                bvu0[i] = b0[i]; bvu0[i + 8] = b1[i];
                bvu1[i] = b2[i]; bvu1[i + 8] = b3[i];
            }
        }
        const v16bf av0 = __builtin_bit_cast(v16bf, avu0);
        const v16bf av1 = __builtin_bit_cast(v16bf, avu1);
        const v16bf bv0 = __builtin_bit_cast(v16bf, bvu0);
        const v16bf bv1 = __builtin_bit_cast(v16bf, bvu1);
        acc00 = __builtin_amdgcn_wmma_f32_16x16x32_bf16(false, av0, false, bv0,
                                                        (short)0, acc00, false, false);
        acc01 = __builtin_amdgcn_wmma_f32_16x16x32_bf16(false, av0, false, bv1,
                                                        (short)0, acc01, false, false);
        acc10 = __builtin_amdgcn_wmma_f32_16x16x32_bf16(false, av1, false, bv0,
                                                        (short)0, acc10, false, false);
        acc11 = __builtin_amdgcn_wmma_f32_16x16x32_bf16(false, av1, false, bv1,
                                                        (short)0, acc11, false, false);

        // ---- convert + mask + store gathered stage s+1 into the other buffer ----
        if (haveNext) {
            unsigned short* Bnx = ldsB[sl & 1];
            #pragma unroll
            for (int j = 0; j < 4; ++j) {
                const unsigned pk = ((unsigned)f2bf(vals[2 * j]) |
                                     ((unsigned)f2bf(vals[2 * j + 1]) << 16)) & okm[j];
                *(unsigned*)&Bnx[(bn0 + 16 * j) * LSTR + 2 * h] = pk;
            }
        }
        asm volatile("s_wait_asynccnt 0x0" ::: "memory");
        __syncthreads();
    }

    // ---- store D: VGPR g -> row m = g + 8*(lane>=16), col n = lane&15 ----
    const int n0 = pos0 + nw * 32 + (lane & 15);
    const int mh = (lane >> 4) * 8;
    float* ob0 = out + ((size_t)b * ON + (o0 + mw * 32)) * HWN;
    float* ob1 = ob0 + (size_t)16 * HWN;
    #pragma unroll
    for (int g = 0; g < 8; ++g) {
        ob0[(size_t)(g + mh) * HWN + n0]      = acc00[g];
        ob0[(size_t)(g + mh) * HWN + n0 + 16] = acc01[g];
        ob1[(size_t)(g + mh) * HWN + n0]      = acc10[g];
        ob1[(size_t)(g + mh) * HWN + n0 + 16] = acc11[g];
    }
}

// ---------------- host launcher ----------------
extern "C" void kernel_launch(void* const* d_in, const int* in_sizes, int n_in,
                              void* d_out, int out_size, void* d_ws, size_t ws_size,
                              hipStream_t stream) {
    (void)in_sizes; (void)n_in; (void)out_size; (void)ws_size;

    const float* x        = (const float*)d_in[0];   // [32,256,56,56]
    const float* proj_w   = (const float*)d_in[1];   // [64,256]
    const float* proj_b   = (const float*)d_in[2];   // [64]
    const float* expert_w = (const float*)d_in[3];   // [8,256,256,3,3]
    float*       out      = (float*)d_out;           // [32,256,56,56]

    // workspace layout (16B-aligned): pooled | r | wmix(bf16, K-permuted)
    float*          pooled = (float*)d_ws;                       // 8192 f32
    float*          r      = pooled + BN * CN;                   // 256 f32
    unsigned short* wmix   = (unsigned short*)(r + BN * EN);     // 32*256*2304 bf16 (~37.7 MB)

    pool_kernel <<<BN * CN, 256, 0, stream>>>(x, pooled);
    route_kernel<<<BN, 64, 0, stream>>>(pooled, proj_w, proj_b, r);
    mix_kernel  <<<ON, 256, 0, stream>>>(expert_w, r, wmix);
    conv_kernel <<<dim3(HWN / 64, ON / 128, BN), 256, 0, stream>>>(x, wmix, out);
}
